// SwinTransformerBlock_53626961658370
// MI455X (gfx1250) — matrix operand order repaired and checked by reading, not compiled, not run on hardware
//
#include <hip/hip_runtime.h>
#include <hip/hip_bf16.h>

// ---------------------------------------------------------------------------
// Swin block on gfx1250: all GEMMs via v_wmma_f32_16x16x32_f16 (fp32 accum),
// A-panel staging via the CDNA5 Tensor Data Mover when the builtin exists.
// B=64 H=W=56 C=96 NH=3 hd=32 WS=7 SS=3 N=49 nW=64 -> T=200704 token rows.
// ---------------------------------------------------------------------------

typedef __attribute__((ext_vector_type(16))) _Float16     v16h;
typedef __attribute__((ext_vector_type(8)))  float        v8f;
typedef __attribute__((ext_vector_type(4)))  unsigned int v4u;
typedef __attribute__((ext_vector_type(8)))  int          v8i;
typedef __attribute__((ext_vector_type(4)))  int          v4i;

union Frag { uint4 q[2]; v16h h; };

#define T_TOK   200704      // 64 * 3136
#define C_DIM   96
#define NHEAD   3
#define HDIM    32
#define NWIN_G  4096        // 64 batches * 64 windows
#define WINN    49
#define HID     384
#define Q_SCALE 0.17677669529663687f   // 32^-0.5

// --- Tensor Data Mover availability (toolchains differ in builtin arity) ---
#if defined(__has_builtin)
#  if __has_builtin(__builtin_amdgcn_tensor_load_to_lds) && \
      __has_builtin(__builtin_amdgcn_s_wait_tensorcnt)
#    define USE_TDM 1
#  endif
#endif
#ifndef USE_TDM
#  define USE_TDM 0
#endif

static __device__ __forceinline__ unsigned int pack2h(float lo, float hi) {
    union { _Float16 h[2]; unsigned int u; } p;
    p.h[0] = (_Float16)lo; p.h[1] = (_Float16)hi;
    return p.u;
}

// ---------------------------------------------------------------------------
// Pre-swizzle weights W[Nout][K] (row-major, fp32) into WMMA B-fragment order:
// dword idx = ((nt*KT + kt)*32 + lane)*8 + v  holds {B[k][n], B[k+1][n]} with
// n = nt*16 + lane%16, k = kt*32 + (lane/16)*16 + 2v, B[k][n] = W[n][k].
// ---------------------------------------------------------------------------
__global__ void pack_weights_kernel(const float* __restrict__ W, unsigned int* __restrict__ pk,
                                    int K, int Nout) {
    int idx = blockIdx.x * 256 + threadIdx.x;
    int total = Nout * K / 2;
    if (idx >= total) return;
    int v    = idx & 7;
    int lane = (idx >> 3) & 31;
    int blk  = idx >> 8;                 // nt*KT + kt
    int KT   = K / 32;
    int kt   = blk % KT;
    int nt   = blk / KT;
    int n    = nt * 16 + (lane & 15);
    int k    = kt * 32 + (lane >> 4) * 16 + 2 * v;
    pk[idx] = pack2h(W[(size_t)n * K + k], W[(size_t)n * K + k + 1]);
}

// ---------------------------------------------------------------------------
// LayerNorm (one wave32 per token row).  mode==1: also apply cyclic shift
// (-3,-3) + 7x7 window partition; output rows are window-ordered.
// ---------------------------------------------------------------------------
__global__ void ln_kernel(const float* __restrict__ in, const float* __restrict__ g,
                          const float* __restrict__ b, _Float16* __restrict__ out, int mode) {
    int wave = threadIdx.x >> 5, lane = threadIdx.x & 31;
    size_t row = (size_t)blockIdx.x * 8 + wave;          // dest row
    const float* src;
    if (mode == 1) {
        int win = (int)(row / WINN), n = (int)(row % WINN);
        int bb = win >> 6, wloc = win & 63;
        int hs = (wloc >> 3) * 7 + n / 7;
        int ws = (wloc & 7) * 7 + n % 7;
        int h = hs + 3; if (h >= 56) h -= 56;
        int w = ws + 3; if (w >= 56) w -= 56;
        src = in + ((size_t)bb * 3136 + h * 56 + w) * C_DIM;
    } else {
        src = in + row * C_DIM;
    }
    float v0 = src[lane], v1 = src[lane + 32], v2 = src[lane + 64];
    float s = v0 + v1 + v2;
    for (int o = 16; o; o >>= 1) s += __shfl_xor(s, o, 32);
    float mean = s * (1.f / 96.f);
    float d0 = v0 - mean, d1 = v1 - mean, d2 = v2 - mean;
    float q = d0 * d0 + d1 * d1 + d2 * d2;
    for (int o = 16; o; o >>= 1) q += __shfl_xor(q, o, 32);
    float rstd = rsqrtf(q * (1.f / 96.f) + 1e-5f);
    _Float16* dst = out + row * C_DIM;
    dst[lane]      = (_Float16)(d0 * rstd * g[lane]      + b[lane]);
    dst[lane + 32] = (_Float16)(d1 * rstd * g[lane + 32] + b[lane + 32]);
    dst[lane + 64] = (_Float16)(d2 * rstd * g[lane + 64] + b[lane + 64]);
}

// ---------------------------------------------------------------------------
// WMMA GEMM: out[M x Ntot] = A[M x K](f16) * B[K x Ntot](pre-swizzled) + bias.
// 256 thr = 8 waves as 4(M) x 2(N); each wave owns MT = MROWS/64 M-subtiles
// sharing one B fragment per k-step (register blocking).  A panel staged to
// LDS by one TDM descriptor (fallback: cooperative b128 copy).
// ---------------------------------------------------------------------------
enum { EPI_QKV = 0, EPI_PROJ = 1, EPI_FC1 = 2, EPI_FC2 = 3 };

template <int K, int MROWS, int EPI>
__global__ __launch_bounds__(256) void
gemm_wmma_kernel(const _Float16* __restrict__ A, const unsigned int* __restrict__ Bpk,
                 const float* __restrict__ bias, int Ntot,
                 float* __restrict__ outF, _Float16* __restrict__ outH,
                 const float* __restrict__ resid,
                 _Float16* __restrict__ Qb, _Float16* __restrict__ Kb,
                 _Float16* __restrict__ Vb) {
    constexpr int KT = K / 32;
    constexpr int MT = MROWS / 64;                 // M-subtiles per wave
    constexpr unsigned NE = (unsigned)MROWS * K;   // panel elements (f16)
    static_assert(NE < 65536, "tile_dim0 must fit 16 bits");
    __shared__ alignas(16) _Float16 As[MROWS * K];

    int tid  = threadIdx.x;
    int wave = tid >> 5, lane = tid & 31;
    int half = lane >> 4, l16 = lane & 15;
    size_t mbase = (size_t)blockIdx.x * MROWS;
    const _Float16* Asrc = A + mbase * K;

#if USE_TDM
    // --- 1-D TDM descriptor: contiguous MROWS*K f16 panel -> LDS ---
    if (wave == 0) {
        unsigned long long ga = (unsigned long long)(const void*)Asrc;
        unsigned lds = (unsigned)(unsigned long long)(const void*)As;
        v4u g0 = { 1u,                                   // count=1 (valid D#)
                   lds,                                  // lds_addr
                   (unsigned)ga,                         // global_addr[31:0]
                   (unsigned)((ga >> 32) & 0x01FFFFFFu) | 0x80000000u }; // [56:32] | type=2
        v8i g1 = { (int)0x00010000u,                     // data_size=1 (2B), mask=0
                   (int)((NE & 0xFFFFu) << 16),          // tensor_dim0 lo16 (<<16)
                   (int)((NE >> 16) | (1u << 16)),       // tensor_dim0 hi16 | tensor_dim1=1
                   (int)(NE << 16),                      // tile_dim0 (bits 127:112)
                   1,                                    // tile_dim1=1, tile_dim2=0
                   (int)NE,                              // tensor_dim0_stride lo32
                   0, 0 };
        v4i gz = { 0, 0, 0, 0 };
#if __clang_major__ >= 23
        v8i gz8 = { 0, 0, 0, 0, 0, 0, 0, 0 };
        __builtin_amdgcn_tensor_load_to_lds(g0, g1, gz, gz, gz8, 0);
#else
        __builtin_amdgcn_tensor_load_to_lds(g0, g1, gz, gz, 0);
#endif
        __builtin_amdgcn_s_wait_tensorcnt(0);
    }
#else
    {   // cooperative coalesced A panel copy
        const uint4* Ag  = reinterpret_cast<const uint4*>(Asrc);
        uint4*       Asv = reinterpret_cast<uint4*>(As);
        constexpr int NV = MROWS * K / 8;
        for (int i = tid; i < NV; i += 256) Asv[i] = Ag[i];
    }
#endif
    __syncthreads();

    int msub = wave >> 1, nsub = wave & 1;
    int nt = blockIdx.y * 2 + nsub;
    int nbase = nt * 16;

    v8f acc[MT];
    float bv = bias[nbase + l16];
    for (int j = 0; j < MT; ++j)
        for (int r = 0; r < 8; ++r) acc[j][r] = bv;

    for (int kt = 0; kt < KT; ++kt) {
        Frag bf;
        const uint4* bv4 = reinterpret_cast<const uint4*>(
            Bpk + (((size_t)nt * KT + kt) * 32 + lane) * 8);
        bf.q[0] = bv4[0];
        bf.q[1] = bv4[1];
        if (kt + 1 < KT)
            __builtin_prefetch(Bpk + (((size_t)nt * KT + kt + 1) * 32 + lane) * 8, 0, 1);
        for (int j = 0; j < MT; ++j) {
            Frag a;
            const _Float16* Arow = As + ((msub * MT + j) * 16 + l16) * K;
            const uint4* av = reinterpret_cast<const uint4*>(Arow + kt * 32 + half * 16);
            a.q[0] = av[0];          // k = kt*32 + half*16 + 0..7
            a.q[1] = av[1];          // k = kt*32 + half*16 + 8..15
            acc[j] = __builtin_amdgcn_wmma_f32_16x16x32_f16(false, a.h, false, bf.h,
                                                            (short)0, acc[j], false, false);
        }
    }

    // epilogue: lane holds D[mbase + (msub*MT+j)*16 + half*8 + r][nbase + l16]
    int c = nbase + l16;
    for (int j = 0; j < MT; ++j) {
        for (int r = 0; r < 8; ++r) {
            int t = (int)mbase + (msub * MT + j) * 16 + half * 8 + r;
            float v = acc[j][r];
            if constexpr (EPI == EPI_QKV) {
                int which = c / 96, hd = (c % 96) / 32, d = c % 32;
                int win = t / WINN, n = t % WINN;
                if (which == 0) v *= Q_SCALE;
                _Float16* dst = (which == 0) ? Qb : (which == 1) ? Kb : Vb;
                dst[(((size_t)win * NHEAD + hd) * 64 + n) * HDIM + d] = (_Float16)v;
            } else if constexpr (EPI == EPI_PROJ) {
                int win = t / WINN, n = t % WINN;
                int bb = win >> 6, wloc = win & 63;
                int hs = (wloc >> 3) * 7 + n / 7;
                int ws = (wloc & 7) * 7 + n % 7;
                int h = hs + 3; if (h >= 56) h -= 56;
                int w = ws + 3; if (w >= 56) w -= 56;
                size_t idx = ((size_t)bb * 3136 + h * 56 + w) * C_DIM + c;
                outF[idx] = v + resid[idx];                   // shortcut + attn-out
            } else if constexpr (EPI == EPI_FC1) {
                float gelu = 0.5f * v * (1.f + erff(v * 0.70710678118654752f));
                outH[(size_t)t * HID + c] = (_Float16)gelu;
            } else {                                          // EPI_FC2
                size_t idx = (size_t)t * C_DIM + c;
                outF[idx] = v + resid[idx];                   // h + mlp
            }
        }
    }
    (void)Ntot;
}

// ---------------------------------------------------------------------------
// Windowed attention, one block per (window, head).  N=49 padded to 64.
// S = (scaled Q)·Kᵀ via 4x4 WMMA tiles (hd=32 == one k-step), softmax in LDS
// fp32 with rel-bias + shift-mask, then P·V via 2 k-steps from LDS.
// ---------------------------------------------------------------------------
__global__ __launch_bounds__(128) void
attn_kernel(const _Float16* __restrict__ Qb, const _Float16* __restrict__ Kb,
            const _Float16* __restrict__ Vb,
            const float* __restrict__ bias_table, const int* __restrict__ rel_index,
            const float* __restrict__ mask, _Float16* __restrict__ out) {
    int blk = blockIdx.x;                  // win*NHEAD + head
    int win = blk / NHEAD, head = blk % NHEAD;
    int tid = threadIdx.x;
    int wave = tid >> 5, lane = tid & 31;
    int half = lane >> 4, l16 = lane & 15;

    const _Float16* Q  = Qb + (size_t)blk * 64 * HDIM;
    const _Float16* Km = Kb + (size_t)blk * 64 * HDIM;
    const _Float16* Vm = Vb + (size_t)blk * 64 * HDIM;

    __shared__ alignas(16) float    S[64 * 64];
    __shared__ alignas(16) _Float16 P[64 * 64];
    __shared__ alignas(16) _Float16 VT[HDIM * 64];

    // stage V transposed: VT[d][j] = V[j][d]
    for (int idx = tid; idx < 64 * HDIM; idx += 128) {
        int j = idx >> 5, d = idx & 31;
        VT[d * 64 + j] = Vm[idx];
    }

    // ---- S = Q·Kᵀ  (wave -> row tile itile) ----
    int itile = wave;
    Frag a;
    {
        const uint4* qv = reinterpret_cast<const uint4*>(Q + (itile * 16 + l16) * HDIM + half * 16);
        a.q[0] = qv[0];
        a.q[1] = qv[1];
    }
    for (int jt = 0; jt < 4; ++jt) {
        Frag bfr;
        const uint4* kv = reinterpret_cast<const uint4*>(Km + (jt * 16 + l16) * HDIM + half * 16);
        bfr.q[0] = kv[0];
        bfr.q[1] = kv[1];
        v8f acc = {};
        acc = __builtin_amdgcn_wmma_f32_16x16x32_f16(false, a.h, false, bfr.h,
                                                     (short)0, acc, false, false);
        for (int r = 0; r < 8; ++r)
            S[(itile * 16 + half * 8 + r) * 64 + jt * 16 + l16] = acc[r];
    }
    __syncthreads();

    // ---- softmax rows (rel bias + shift mask), P in f16, pad cols zeroed ----
    if (tid < 64) {
        int i = tid;
        _Float16* prow = P + i * 64;
        if (i < WINN) {
            float* srow = S + i * 64;
            const float* mrow = mask + ((size_t)(win & 63)) * WINN * WINN + i * WINN;
            const int*   ridx = rel_index + i * WINN;
            float mx = -1e30f;
            for (int j = 0; j < WINN; ++j) {
                float v = srow[j] + bias_table[ridx[j] * NHEAD + head] + mrow[j];
                srow[j] = v;
                mx = fmaxf(mx, v);
            }
            float sum = 0.f;
            for (int j = 0; j < WINN; ++j) {
                float e = __expf(srow[j] - mx);
                srow[j] = e;
                sum += e;
            }
            float inv = 1.f / sum;
            for (int j = 0; j < WINN; ++j) prow[j] = (_Float16)(srow[j] * inv);
            for (int j = WINN; j < 64; ++j) prow[j] = (_Float16)0.f;
        } else {
            for (int j = 0; j < 64; ++j) prow[j] = (_Float16)0.f;
        }
    }
    __syncthreads();

    // ---- O = P·V  (K dim = 64 -> two 32-steps) ----
    for (int dsub = 0; dsub < 2; ++dsub) {
        v8f acc = {};
        for (int ks = 0; ks < 2; ++ks) {
            Frag pa, vb;
            const uint4* pv = reinterpret_cast<const uint4*>(
                P + (itile * 16 + l16) * 64 + ks * 32 + half * 16);
            pa.q[0] = pv[0];
            pa.q[1] = pv[1];
            const uint4* vv = reinterpret_cast<const uint4*>(
                VT + (dsub * 16 + l16) * 64 + ks * 32 + half * 16);
            vb.q[0] = vv[0];
            vb.q[1] = vv[1];
            acc = __builtin_amdgcn_wmma_f32_16x16x32_f16(false, pa.h, false, vb.h,
                                                         (short)0, acc, false, false);
        }
        for (int r = 0; r < 8; ++r) {
            int i = itile * 16 + half * 8 + r;
            if (i < WINN) {
                size_t t = (size_t)win * WINN + i;
                out[t * C_DIM + head * HDIM + dsub * 16 + l16] = (_Float16)acc[r];
            }
        }
    }
}

// ---------------------------------------------------------------------------
extern "C" void kernel_launch(void* const* d_in, const int* in_sizes, int n_in,
                              void* d_out, int out_size, void* d_ws, size_t ws_size,
                              hipStream_t stream) {
    (void)in_sizes; (void)n_in; (void)out_size; (void)ws_size;

    const float* x         = (const float*)d_in[0];
    const float* attn_mask = (const float*)d_in[1];
    const float* ln1_g     = (const float*)d_in[2];
    const float* ln1_b     = (const float*)d_in[3];
    const float* qkv_w     = (const float*)d_in[4];
    const float* qkv_b     = (const float*)d_in[5];
    const float* rel_tab   = (const float*)d_in[6];
    const int*   rel_idx   = (const int*)d_in[7];
    const float* proj_w    = (const float*)d_in[8];
    const float* proj_b    = (const float*)d_in[9];
    const float* ln2_g     = (const float*)d_in[10];
    const float* ln2_b     = (const float*)d_in[11];
    const float* fc1_w     = (const float*)d_in[12];
    const float* fc1_b     = (const float*)d_in[13];
    const float* fc2_w     = (const float*)d_in[14];
    const float* fc2_b     = (const float*)d_in[15];
    float* outp = (float*)d_out;

    // workspace carve-up (256B aligned)
    char* p = (char*)d_ws;
    auto carve = [&](size_t n) { char* r = p; p += (n + 255) & ~(size_t)255; return r; };
    _Float16* xw     = (_Float16*)carve((size_t)T_TOK * C_DIM * 2);
    _Float16* Qbuf   = (_Float16*)carve((size_t)NWIN_G * NHEAD * 64 * HDIM * 2);
    _Float16* Kbuf   = (_Float16*)carve((size_t)NWIN_G * NHEAD * 64 * HDIM * 2);
    _Float16* Vbuf   = (_Float16*)carve((size_t)NWIN_G * NHEAD * 64 * HDIM * 2);
    _Float16* attn_o = (_Float16*)carve((size_t)T_TOK * C_DIM * 2);
    float*    h_buf  = (float*)   carve((size_t)T_TOK * C_DIM * 4);
    _Float16* m_buf  = (_Float16*)carve((size_t)T_TOK * C_DIM * 2);
    _Float16* fc1_o  = (_Float16*)carve((size_t)T_TOK * HID * 2);
    unsigned int* pk_qkv  = (unsigned int*)carve((size_t)288 * 96 * 2);
    unsigned int* pk_proj = (unsigned int*)carve((size_t)96  * 96 * 2);
    unsigned int* pk_fc1  = (unsigned int*)carve((size_t)384 * 96 * 2);
    unsigned int* pk_fc2  = (unsigned int*)carve((size_t)96  * 384 * 2);

    // 0) pre-swizzle weights into WMMA B-fragment order
    pack_weights_kernel<<<(288 * 96 / 2 + 255) / 256, 256, 0, stream>>>(qkv_w, pk_qkv, 96, 288);
    pack_weights_kernel<<<(96 * 96 / 2 + 255) / 256, 256, 0, stream>>>(proj_w, pk_proj, 96, 96);
    pack_weights_kernel<<<(384 * 96 / 2 + 255) / 256, 256, 0, stream>>>(fc1_w, pk_fc1, 96, 384);
    pack_weights_kernel<<<(96 * 384 / 2 + 255) / 256, 256, 0, stream>>>(fc2_w, pk_fc2, 384, 96);

    // 1) LN1 + shift + window partition -> xw (f16, window-ordered rows)
    ln_kernel<<<T_TOK / 8, 256, 0, stream>>>(x, ln1_g, ln1_b, xw, 1);

    // 2) QKV: [T,96]x[96,288], split into padded Q/K/V (q pre-scaled)
    gemm_wmma_kernel<96, 128, EPI_QKV><<<dim3(T_TOK / 128, 288 / 32), 256, 0, stream>>>(
        xw, pk_qkv, qkv_b, 288, nullptr, nullptr, nullptr, Qbuf, Kbuf, Vbuf);

    // 3) windowed attention per (window, head)
    attn_kernel<<<NWIN_G * NHEAD, 128, 0, stream>>>(Qbuf, Kbuf, Vbuf, rel_tab, rel_idx,
                                                    attn_mask, attn_o);

    // 4) proj + window reverse + unshift + residual -> h_buf (fp32)
    gemm_wmma_kernel<96, 128, EPI_PROJ><<<dim3(T_TOK / 128, 96 / 32), 256, 0, stream>>>(
        attn_o, pk_proj, proj_b, 96, h_buf, nullptr, x, nullptr, nullptr, nullptr);

    // 5) LN2 -> m (f16)
    ln_kernel<<<T_TOK / 8, 256, 0, stream>>>(h_buf, ln2_g, ln2_b, m_buf, 0);

    // 6) FC1 + exact GELU -> fc1_o (f16)
    gemm_wmma_kernel<96, 128, EPI_FC1><<<dim3(T_TOK / 128, HID / 32), 256, 0, stream>>>(
        m_buf, pk_fc1, fc1_b, HID, nullptr, fc1_o, nullptr, nullptr, nullptr, nullptr);

    // 7) FC2 + residual -> d_out (fp32)  (64-row tiles: 48KB LDS panel)
    gemm_wmma_kernel<384, 64, EPI_FC2><<<dim3(T_TOK / 64, 96 / 32), 256, 0, stream>>>(
        fc1_o, pk_fc2, fc2_b, 96, outp, nullptr, h_buf, nullptr, nullptr, nullptr);
}
